// preCHANNEL_72662256714427
// MI455X (gfx1250) — matrix operand order repaired and checked by reading, not compiled, not run on hardware
//
#include <hip/hip_runtime.h>
#include <hip/hip_bf16.h>
#include <stdint.h>

// ---------------------------------------------------------------------------
// Types for CDNA5 WMMA (wave32): bf16 A/B fragments, f32 accumulator.
// ---------------------------------------------------------------------------
typedef __bf16 bf16_t;
typedef bf16_t v16bf __attribute__((ext_vector_type(16)));
typedef float  v8f   __attribute__((ext_vector_type(8)));

__device__ __forceinline__ unsigned short f2bf(float f) {
  unsigned int u = __float_as_uint(f);
  u += 0x7FFFu + ((u >> 16) & 1u);        // round-to-nearest-even
  return (unsigned short)(u >> 16);
}
__device__ __forceinline__ float bf2f(unsigned short h) {
  return __uint_as_float(((unsigned int)h) << 16);
}

// ---------------------------------------------------------------------------
// CDNA5 async global->LDS copy (ASYNCcnt), 16B per lane.
//   Probe-confirmed: builtin exists, param0 = int4-vector ptr in AS(1).
// ---------------------------------------------------------------------------
typedef int v4i_gcc __attribute__((vector_size(16)));
typedef __attribute__((address_space(1))) v4i_gcc g_v4i;
typedef __attribute__((address_space(3))) v4i_gcc l_v4i;

__device__ __forceinline__ void asyncCopy16(const unsigned short* g,
                                            unsigned short* l) {
#if __has_builtin(__builtin_amdgcn_global_load_async_to_lds_b128)
  __builtin_amdgcn_global_load_async_to_lds_b128(
      (g_v4i*)g, (l_v4i*)l, /*offset=*/0, /*cpol=*/0);
#else
  unsigned int loff = (unsigned int)(unsigned long long)
      (__attribute__((address_space(3))) unsigned short*)l;
  asm volatile("global_load_async_to_lds_b128 %0, %1, off"
               :: "v"(loff), "v"((unsigned long long)(size_t)g)
               : "memory");
#endif
}
__device__ __forceinline__ void waitAsync() {
#if __has_builtin(__builtin_amdgcn_s_wait_asynccnt)
  __builtin_amdgcn_s_wait_asynccnt(0);
#else
  asm volatile("s_wait_asynccnt 0" ::: "memory");
#endif
}

// ---------------------------------------------------------------------------
// Input convert: X f32 NCHW [N,3,H,W] -> bf16 NHWC [N,H,W,3]
// ---------------------------------------------------------------------------
__global__ void cvt_in_kernel(const float* __restrict__ in,
                              unsigned short* __restrict__ out,
                              int HW, int n) {            // n = N*HW*3
  int i = blockIdx.x * blockDim.x + threadIdx.x;
  if (i >= n) return;
  int c = i % 3;
  int t = i / 3;
  int img = t / HW;
  int rem = t - img * HW;
  out[i] = f2bf(in[((size_t)img * 3 + c) * HW + rem]);
}

// ---------------------------------------------------------------------------
// Weight pack: w f32 [Cout,Cin,3,3] -> wp bf16 [Cout,Kpad],
//              k = (r*3+s)*Cin + ci  (rs-major), zero padded to Kpad
// ---------------------------------------------------------------------------
__global__ void pack_w_kernel(const float* __restrict__ w,
                              unsigned short* __restrict__ wp,
                              int Cin, int Cout, int Kreal, int Kpad) {
  int i = blockIdx.x * blockDim.x + threadIdx.x;
  if (i >= Cout * Kpad) return;
  int row = i / Kpad;
  int k   = i - row * Kpad;
  unsigned short v = 0;
  if (k < Kreal) {
    int rs = k / Cin;
    int ci = k - rs * Cin;
    v = f2bf(w[((size_t)row * Cin + ci) * 9 + rs]);
  }
  wp[i] = v;
}

// ---------------------------------------------------------------------------
// Implicit-GEMM 3x3 SAME conv + bias + ReLU, bf16 NHWC in/out, f32 accumulate.
//   Block tile 64ch x 256px, K-step 32. 8 waves in 2(m) x 4(n) grid;
//   wave tile 32ch x 64px -> 8 WMMA per K-step.
//   Fast path: double-buffered LDS, 1 barrier/step, async global->LDS staging
//   issued one step ahead (s_wait_asynccnt after the WMMA block).
// ---------------------------------------------------------------------------
#define BM 64
#define BN 256
#define BK 32
#define LDSST 40   // LDS row stride in halves (80B) -> conflict-free fragments

__global__ __launch_bounds__(256)
void conv3x3_wmma_kernel(const unsigned short* __restrict__ in,   // bf16 NHWC
                         const unsigned short* __restrict__ wp,   // bf16 [Cout,Kpad]
                         const float* __restrict__ bias,
                         unsigned short* __restrict__ out,        // bf16 NHWC
                         int Cin, int Cout, int H, int W, int Kpad, int Nimg) {
  __shared__ __align__(16) unsigned short Asq[2][BM * LDSST]; // [m][k]
  __shared__ __align__(16) unsigned short Bsq[2][BN * LDSST]; // [pix][k]

  const int HW       = H * W;
  const int totalPix = Nimg * HW;
  const int pixBase  = blockIdx.x * BN;
  const int coBase   = blockIdx.y * BM;
  const int tid      = threadIdx.x;
  const int lane     = tid & 31;
  const int wave     = tid >> 5;
  const int waveM    = wave >> 2;   // 0..1 : 32-channel sub-tile
  const int waveN    = wave & 3;    // 0..3 : 64-pixel sub-tile
  const int laneLow  = lane & 15;
  const int laneHi   = lane >> 4;

  // ---- B-fill: thread owns one pixel, all 32 k's of the step ----
  const int p = pixBase + tid;
  const bool pvalid = (p < totalPix);
  int pn = 0, py = 0, px = 0;
  if (pvalid) { pn = p / HW; int rem = p - pn * HW; py = rem / W; px = rem - py * W; }
  const int bDst = tid * LDSST;

  // ---- A-fill: thread -> (m = tid>>2, 8 k's at (tid&3)*8) ----
  const size_t wrow = (size_t)(coBase + (tid >> 2)) * Kpad + (size_t)((tid & 3) * 8);
  const int    aDst = (tid >> 2) * LDSST + (tid & 3) * 8;

  const int mrow  = (waveM * 32 + laneLow) * LDSST;
  const int nrow0 = (waveN * 64 + laneLow) * LDSST;

  v8f acc[2][4] = {};
  const uint4 zero4 = {0u, 0u, 0u, 0u};

  // one K-step: preload all 6 fragments (12x ds_load_b128, one dscnt wait),
  // then 8 back-to-back WMMAs on distinct registers.
  auto computeStep = [&](int b) {
    union { uint4 q[2]; v16bf v; } afr[2], bfr[4];
#pragma unroll
    for (int mi = 0; mi < 2; mi++) {
      const int mr = mrow + mi * 16 * LDSST;
      afr[mi].q[0] = *(const uint4*)(&Asq[b][mr + laneHi * 8]);
      afr[mi].q[1] = *(const uint4*)(&Asq[b][mr + 16 + laneHi * 8]);
    }
#pragma unroll
    for (int j = 0; j < 4; j++) {
      const int nr = nrow0 + j * 16 * LDSST;
      bfr[j].q[0] = *(const uint4*)(&Bsq[b][nr + laneHi * 8]);
      bfr[j].q[1] = *(const uint4*)(&Bsq[b][nr + 16 + laneHi * 8]);
    }
#pragma unroll
    for (int j = 0; j < 4; j++) {
      acc[0][j] = __builtin_amdgcn_wmma_f32_16x16x32_bf16(
          false, afr[0].v, false, bfr[j].v, (short)0, acc[0][j], false, false);
      acc[1][j] = __builtin_amdgcn_wmma_f32_16x16x32_bf16(
          false, afr[1].v, false, bfr[j].v, (short)0, acc[1][j], false, false);
    }
  };

  if ((Cin & 31) == 0) {
    // ================= fast path: async-staged, double-buffered =============
    const unsigned short* srcS = in;
    bool okS = false;
    auto rsSetup = [&](int rs) {
      const int r = (rs >= 6) ? 2 : (rs >= 3 ? 1 : 0);
      const int s = rs - r * 3;
      const int yy = py + r - 1;
      const int xx = px + s - 1;
      okS = pvalid && ((unsigned)yy < (unsigned)H) && ((unsigned)xx < (unsigned)W);
      srcS = in + ((size_t)(pn * H + yy) * W + xx) * Cin;
    };

    // stage step (rsS*Cin + c0S) into buffer nb
    auto stage = [&](int nb, int k0, int c0) {
      asyncCopy16(wp + wrow + k0, &Asq[nb][aDst]);
      if (okS) {
        asyncCopy16(srcS + c0,      &Bsq[nb][bDst]);
        asyncCopy16(srcS + c0 + 8,  &Bsq[nb][bDst + 8]);
        asyncCopy16(srcS + c0 + 16, &Bsq[nb][bDst + 16]);
        asyncCopy16(srcS + c0 + 24, &Bsq[nb][bDst + 24]);
      } else {
        *(uint4*)(&Bsq[nb][bDst])      = zero4;
        *(uint4*)(&Bsq[nb][bDst + 8])  = zero4;
        *(uint4*)(&Bsq[nb][bDst + 16]) = zero4;
        *(uint4*)(&Bsq[nb][bDst + 24]) = zero4;
      }
    };

    const int nsteps = 9 * (Cin >> 5);
    int rsS = 0, c0S = 0;
    rsSetup(0);
    stage(0, 0, 0);                           // prologue: stage step 0
    waitAsync();

    for (int t = 0; t < nsteps; t++) {
      __syncthreads();                        // buffer t&1 now complete
      const bool hasNext = (t + 1) < nsteps;  // uniform
      if (hasNext) {
        c0S += BK;
        if (c0S == Cin) { c0S = 0; rsS++; rsSetup(rsS); }   // uniform branch
        stage((t + 1) & 1, rsS * Cin + c0S, c0S);
      }
      computeStep(t & 1);                     // async copies run under WMMAs
      waitAsync();                            // drain before next barrier
    }
  } else {
    // ================= generic path (first layer, Cin=3) ====================
    const int Kreal = Cin * 9;
    for (int k0 = 0; k0 < Kpad; k0 += BK) {
      uint4 av = *(const uint4*)(wp + wrow + k0);
      *(uint4*)(&Asq[0][aDst]) = av;
#pragma unroll 4
      for (int e = 0; e < BK; e++) {
        int kg = k0 + e;
        unsigned short val = 0;
        if (pvalid && kg < Kreal) {
          int rs = kg / Cin;
          int ci = kg - rs * Cin;
          int r  = rs / 3;
          int s  = rs - r * 3;
          int yy = py + r - 1;
          int xx = px + s - 1;
          if ((unsigned)yy < (unsigned)H && (unsigned)xx < (unsigned)W)
            val = in[((size_t)(pn * H + yy) * W + xx) * Cin + ci];
        }
        Bsq[0][bDst + e] = val;
      }
      __syncthreads();
      computeStep(0);
      __syncthreads();
    }
  }

  // ---- epilogue: lane holds 8 consecutive channels -> b128 stores ----
#pragma unroll
  for (int mi = 0; mi < 2; mi++) {
    const int coB = coBase + waveM * 32 + mi * 16 + laneHi * 8; // multiple of 8
    const float4 bb0 = *(const float4*)(bias + coB);
    const float4 bb1 = *(const float4*)(bias + coB + 4);
    const float bv[8] = {bb0.x, bb0.y, bb0.z, bb0.w, bb1.x, bb1.y, bb1.z, bb1.w};
#pragma unroll
    for (int j = 0; j < 4; j++) {
      int p2 = pixBase + waveN * 64 + j * 16 + laneLow;
      if (p2 >= totalPix) continue;
      unsigned int ww[4];
#pragma unroll
      for (int h = 0; h < 4; h++) {
        float v0 = acc[mi][j][2 * h]     + bv[2 * h];
        float v1 = acc[mi][j][2 * h + 1] + bv[2 * h + 1];
        v0 = v0 > 0.f ? v0 : 0.f;
        v1 = v1 > 0.f ? v1 : 0.f;
        ww[h] = (unsigned int)f2bf(v0) | ((unsigned int)f2bf(v1) << 16);
      }
      uint4 st = {ww[0], ww[1], ww[2], ww[3]};
      *(uint4*)(out + (size_t)p2 * Cout + coB) = st;
    }
  }
}

// ---------------------------------------------------------------------------
// 2x2 stride-2 maxpool, bf16 NHWC; thread handles 8 consecutive channels.
// ---------------------------------------------------------------------------
__global__ void maxpool2_kernel(const unsigned short* __restrict__ in,
                                unsigned short* __restrict__ out,
                                int C, int W, int total8) {
  int i = blockIdx.x * blockDim.x + threadIdx.x;
  if (i >= total8) return;
  const int C8 = C >> 3;
  int c8 = i % C8;
  int t  = i / C8;                 // (n*Ho + yo)*Wo + xo
  const int Wo = W >> 1;
  int xo = t % Wo;
  int t2 = t / Wo;                 // n*Ho + yo ; input row = 2*t2 (H = 2*Ho)
  size_t base = ((size_t)(2 * t2) * W + 2 * xo) * C + (size_t)c8 * 8;
  const size_t sx = C;
  const size_t sy = (size_t)W * C;
  uint4 q0 = *(const uint4*)(in + base);
  uint4 q1 = *(const uint4*)(in + base + sx);
  uint4 q2 = *(const uint4*)(in + base + sy);
  uint4 q3 = *(const uint4*)(in + base + sy + sx);
  const unsigned short* a = (const unsigned short*)&q0;
  const unsigned short* b = (const unsigned short*)&q1;
  const unsigned short* c = (const unsigned short*)&q2;
  const unsigned short* d = (const unsigned short*)&q3;
  unsigned short r[8];
#pragma unroll
  for (int e = 0; e < 8; e++) {
    float m = fmaxf(fmaxf(bf2f(a[e]), bf2f(b[e])),
                    fmaxf(bf2f(c[e]), bf2f(d[e])));
    r[e] = f2bf(m);
  }
  *(uint4*)(out + (size_t)i * 8) = *(uint4*)r;
}

// ---------------------------------------------------------------------------
// Argmax over 28x28 per (n,c), NHWC feat [4,28,28,512]; thread = (n,c).
// First-max semantics (ascending scan, strict >). out[c][n] = (row, col) int32.
// ---------------------------------------------------------------------------
__global__ void argmax_kernel(const unsigned short* __restrict__ feat,
                              int* __restrict__ out, int Nimg) {
  int idx = blockIdx.x * blockDim.x + threadIdx.x;
  if (idx >= Nimg * 512) return;
  int n = idx >> 9;
  int c = idx & 511;
  const unsigned short* p = feat + (size_t)n * 784 * 512 + c;
  float bv = -3.4e38f;
  int bi = 0;
  for (int i = 0; i < 784; i++) {
    float v = bf2f(p[(size_t)i * 512]);
    if (v > bv) { bv = v; bi = i; }
  }
  out[(c * Nimg + n) * 2 + 0] = bi / 28;
  out[(c * Nimg + n) * 2 + 1] = bi % 28;
}

// ---------------------------------------------------------------------------
// Host orchestration (graph-capture safe: launches only, all on stream).
// d_in: [0]=X, then (w,b) x 13 -> w_l = d_in[1+2l], b_l = d_in[2+2l].
// ---------------------------------------------------------------------------
extern "C" void kernel_launch(void* const* d_in, const int* in_sizes, int n_in,
                              void* d_out, int out_size, void* d_ws, size_t ws_size,
                              hipStream_t stream) {
  (void)in_sizes; (void)n_in; (void)out_size; (void)ws_size;
  static const int cin_[13]  = {3, 64, 64, 128, 128, 256, 256, 256, 512, 512, 512, 512, 512};
  static const int cout_[13] = {64, 64, 128, 128, 256, 256, 256, 512, 512, 512, 512, 512, 512};
  static const int cfg_[17]  = {64, 64, -1, 128, 128, -1, 256, 256, 256, -1,
                                512, 512, 512, -1, 512, 512, 512};
  const int Nimg = 4;

  // -------- workspace layout --------
  char* ws = (char*)d_ws;
  size_t off = 0;
  unsigned short* wp[13];
  for (int l = 0; l < 13; l++) {
    int Kreal = cin_[l] * 9;
    int Kpad  = (Kreal + 31) & ~31;
    wp[l] = (unsigned short*)(ws + off);
    off += (size_t)cout_[l] * Kpad * 2;
    off = (off + 255) & ~(size_t)255;
  }
  const size_t bufElems = (size_t)Nimg * 64 * 448 * 448;  // largest activation
  unsigned short* bufA = (unsigned short*)(ws + off);
  off += bufElems * 2; off = (off + 255) & ~(size_t)255;
  unsigned short* bufB = (unsigned short*)(ws + off);

  // -------- pack all conv weights to bf16 [Cout, Kpad] (rs-major K) --------
  for (int l = 0; l < 13; l++) {
    int Kreal = cin_[l] * 9;
    int Kpad  = (Kreal + 31) & ~31;
    int tot   = cout_[l] * Kpad;
    pack_w_kernel<<<(tot + 255) / 256, 256, 0, stream>>>(
        (const float*)d_in[1 + 2 * l], wp[l], cin_[l], cout_[l], Kreal, Kpad);
  }

  // -------- convert input image to bf16 NHWC --------
  {
    int n = Nimg * 448 * 448 * 3;
    cvt_in_kernel<<<(n + 255) / 256, 256, 0, stream>>>(
        (const float*)d_in[0], bufA, 448 * 448, n);
  }

  // -------- run the conv/pool stack, ping-ponging bufA/bufB --------
  unsigned short* cur = bufA;
  unsigned short* nxt = bufB;
  int H = 448, C = 3, li = 0;
  for (int s = 0; s < 17; s++) {
    if (cfg_[s] < 0) {
      int Ho = H / 2;
      int tot8 = Nimg * Ho * Ho * (C / 8);
      maxpool2_kernel<<<(tot8 + 255) / 256, 256, 0, stream>>>(cur, nxt, C, H, tot8);
      H = Ho;
    } else {
      int Cout  = cfg_[s];
      int Kreal = C * 9;
      int Kpad  = (Kreal + 31) & ~31;
      int totalPix = Nimg * H * H;
      dim3 grid((totalPix + BN - 1) / BN, Cout / BM);
      conv3x3_wmma_kernel<<<grid, 256, 0, stream>>>(
          cur, wp[li], (const float*)d_in[2 + 2 * li], nxt,
          C, Cout, H, H, Kpad, Nimg);
      C = Cout;
      li++;
    }
    unsigned short* t = cur; cur = nxt; nxt = t;
  }

  // -------- argmax over 28x28 per (n,c) --------
  argmax_kernel<<<(Nimg * 512 + 255) / 256, 256, 0, stream>>>(
      cur, (int*)d_out, Nimg);
}
